// Net_33990371180587
// MI455X (gfx1250) — matrix operand order repaired and checked by reading, not compiled, not run on hardware
//
#include <hip/hip_runtime.h>
#include <hip/hip_bf16.h>

typedef __attribute__((ext_vector_type(16))) __bf16   v16bf;
typedef __attribute__((ext_vector_type(8)))  __bf16   v8bf;
typedef __attribute__((ext_vector_type(8)))  float    v8f;
typedef __attribute__((ext_vector_type(4)))  unsigned su4;
typedef __attribute__((ext_vector_type(8)))  unsigned su8;

#define BT    1024      // B*T
#define IDIM  256
#define HDIM  1024
#define ODIM  256

// ---------------------------------------------------------------------------
// CDNA5 async / TDM helpers
// ---------------------------------------------------------------------------
__device__ __forceinline__ unsigned lds_addr_of(const void* p) {
    // generic (flat) LDS pointer: low 32 bits are the LDS byte address
    return (unsigned)(size_t)p;
}

__device__ __forceinline__ void async_copy_b128(unsigned lds_off, const void* gptr) {
    // per-lane 16B global->LDS copy, tracked on ASYNCcnt
    asm volatile("global_load_async_to_lds_b128 %0, %1, off"
                 :: "v"(lds_off), "v"(gptr) : "memory");
}

__device__ __forceinline__ void wait_async_all() {
    asm volatile("s_wait_asynccnt 0x0" ::: "memory");
}

// ---------------------------------------------------------------------------
// Kernel 1: argaug — one block (256 thr) per (b,t) row.
// sim(s) = dot(y, win_s) * rsqrt(||win_s||^2)  (||y|| constant per row -> drop)
// ---------------------------------------------------------------------------
__global__ __launch_bounds__(256) void argaug_kernel(
    const float* __restrict__ x, const float* __restrict__ y,
    float* __restrict__ x_aug, __bf16* __restrict__ x_aug_bf)
{
    __shared__ float s_x[IDIM];
    __shared__ float s_y[IDIM];
    __shared__ float s_key[896];          // max klen = 384 + 510 = 894
    __shared__ float s_rsim[256];
    __shared__ int   s_rth[256];
    __shared__ int   s_rs[256];
    __shared__ int   s_bth, s_bs;

    const int row = blockIdx.x;
    const int tid = threadIdx.x;

    s_x[tid] = x[row * IDIM + tid];
    s_y[tid] = y[row * IDIM + tid];
    __syncthreads();

    float best_sim = -3.0e38f;
    int   best_th  = 0;
    int   best_s   = 0;

    for (int th = 0; th < 9; ++th) {
        const int idim_k = 128 + 32 * th;           // floor(256*theta), exact
        const int klen   = idim_k + 510;            // 255 zero pad both sides
        for (int i = tid; i < klen; i += 256) {
            const int j = i - 255;
            float v = 0.0f;
            if (j >= 0 && j < idim_k) v = s_x[(j * IDIM) / idim_k];
            s_key[i] = v;
        }
        __syncthreads();

        const int n_shift = idim_k + 255;
        for (int s = tid; s < n_shift; s += 256) {
            float dot = 0.0f, nrm = 0.0f;
            #pragma unroll 8
            for (int q = 0; q < IDIM; ++q) {
                const float k = s_key[s + q];       // consecutive lanes: no conflict
                dot = fmaf(s_y[q], k, dot);         // s_y[q]: broadcast
                nrm = fmaf(k, k, nrm);
            }
            const float sim = dot * rsqrtf(nrm);
            if (sim > best_sim) { best_sim = sim; best_th = th; best_s = s; }
        }
        __syncthreads();
    }

    // block-wide argmax with reference tie-breaking (higher sim, lower theta, lower s)
    s_rsim[tid] = best_sim; s_rth[tid] = best_th; s_rs[tid] = best_s;
    __syncthreads();
    if (tid == 0) {
        float bs = s_rsim[0]; int bt = s_rth[0]; int bp = s_rs[0];
        for (int i = 1; i < 256; ++i) {
            const float v = s_rsim[i];
            if (v > bs || (v == bs && (s_rth[i] < bt || (s_rth[i] == bt && s_rs[i] < bp)))) {
                bs = v; bt = s_rth[i]; bp = s_rs[i];
            }
        }
        s_bth = bt; s_bs = bp;
    }
    __syncthreads();

    // rebuild winning key_pad and emit the winning window
    const int bth = s_bth, bsh = s_bs;
    const int idim_k = 128 + 32 * bth;
    const int klen   = idim_k + 510;
    for (int i = tid; i < klen; i += 256) {
        const int j = i - 255;
        float v = 0.0f;
        if (j >= 0 && j < idim_k) v = s_x[(j * IDIM) / idim_k];
        s_key[i] = v;
    }
    __syncthreads();
    const float v = s_key[bsh + tid];
    x_aug[row * IDIM + tid]    = v;
    x_aug_bf[row * IDIM + tid] = (__bf16)v;
}

// ---------------------------------------------------------------------------
// Kernel 2: transpose+convert weights to bf16 column-major (contiguous-K frags)
// ---------------------------------------------------------------------------
__global__ __launch_bounds__(256) void convert_weights_kernel(
    const float* __restrict__ W1, const float* __restrict__ W2,
    __bf16* __restrict__ W1t, __bf16* __restrict__ W2t)
{
    const unsigned idx = blockIdx.x * 256u + threadIdx.x;
    if (idx < IDIM * HDIM) {                        // W1: [256 x 1024] -> W1t [1024 x 256]
        const unsigned k = idx >> 10, n = idx & 1023u;
        W1t[n * IDIM + k] = (__bf16)W1[k * HDIM + n];
    } else if (idx < 2u * IDIM * HDIM) {            // W2: [1024 x 256] -> W2t [256 x 1024]
        const unsigned j = idx - IDIM * HDIM;
        const unsigned k = j >> 8, n = j & 255u;
        W2t[n * HDIM + k] = (__bf16)W2[k * ODIM + n];
    }
}

// ---------------------------------------------------------------------------
// Kernel 3: JAX threefry2x32-20 Bernoulli mask, key=(0,42), counts split-halves
// ---------------------------------------------------------------------------
__device__ __forceinline__ unsigned rotl32(unsigned v, int n) {
    return (v << n) | (v >> (32 - n));
}

__global__ __launch_bounds__(256) void mask_kernel(
    const float* __restrict__ freq, unsigned char* __restrict__ mk)
{
    const unsigned i = blockIdx.x * 256u + threadIdx.x;
    const unsigned HALF = (BT * HDIM) / 2u;         // 524288
    if (i >= HALF) return;

    const unsigned k0 = 0u, k1 = 42u;
    const unsigned k2 = k0 ^ k1 ^ 0x1BD11BDAu;
    unsigned x0 = i, x1 = i + HALF;

    x0 += k0; x1 += k1;
    { const int R[4] = {13,15,26,6};
      #pragma unroll
      for (int r = 0; r < 4; ++r) { x0 += x1; x1 = rotl32(x1, R[r]); x1 ^= x0; } }
    x0 += k1; x1 += k2 + 1u;
    { const int R[4] = {17,29,16,24};
      #pragma unroll
      for (int r = 0; r < 4; ++r) { x0 += x1; x1 = rotl32(x1, R[r]); x1 ^= x0; } }
    x0 += k2; x1 += k0 + 2u;
    { const int R[4] = {13,15,26,6};
      #pragma unroll
      for (int r = 0; r < 4; ++r) { x0 += x1; x1 = rotl32(x1, R[r]); x1 ^= x0; } }
    x0 += k0; x1 += k1 + 3u;
    { const int R[4] = {17,29,16,24};
      #pragma unroll
      for (int r = 0; r < 4; ++r) { x0 += x1; x1 = rotl32(x1, R[r]); x1 ^= x0; } }
    x0 += k1; x1 += k2 + 4u;

    const float u0 = __uint_as_float((x0 >> 9) | 0x3f800000u) - 1.0f;
    const float u1 = __uint_as_float((x1 >> 9) | 0x3f800000u) - 1.0f;
    mk[i]        = (u0 < freq[i & (HDIM - 1u)]) ? 1u : 0u;
    mk[i + HALF] = (u1 < freq[(i + HALF) & (HDIM - 1u)]) ? 1u : 0u;
}

// ---------------------------------------------------------------------------
// WMMA fragment loaders (ISA 7.12.2: 16-bit A 16x32 / B 32x16 VGPR layouts)
// Work on generic pointers: after inlining on __shared__ arrays these lower
// to ds_load_b128 (16B-aligned by construction).
// ---------------------------------------------------------------------------
__device__ __forceinline__ v16bf load_fragA(const __bf16* base,
                                            int ld, int m0, int k0, int lane)
{
    // lane<16: row m0+lane, K {k0..k0+7} U {k0+16..k0+23}; lane>=16: +8 offset
    const int r  = m0 + (lane & 15);
    const int kb = k0 + ((lane >> 4) << 3);
    const v8bf lo = *(const v8bf*)(base + r * ld + kb);
    const v8bf hi = *(const v8bf*)(base + r * ld + kb + 16);
    v16bf out;
    #pragma unroll
    for (int j = 0; j < 8; ++j) { out[j] = lo[j]; out[8 + j] = hi[j]; }
    return out;
}

__device__ __forceinline__ v16bf load_fragB(const __bf16* baseT,
                                            int ldk, int n0, int k0, int lane)
{
    // B stored transposed [N][K]; lane<16: col n0+lane holds K {k0..k0+7} U {k0+16..k0+23},
    // lane>=16: col n0+lane-16 holds K {k0+8..k0+15} U {k0+24..k0+31}
    const int n  = n0 + (lane & 15);
    const int kb = k0 + ((lane >> 4) << 3);
    const v8bf lo = *(const v8bf*)(baseT + n * ldk + kb);
    const v8bf hi = *(const v8bf*)(baseT + n * ldk + kb + 16);
    v16bf out;
    #pragma unroll
    for (int j = 0; j < 8; ++j) { out[j] = lo[j]; out[8 + j] = hi[j]; }
    return out;
}

// ---------------------------------------------------------------------------
// Kernel 4: GEMM1  H = bf16( (x_aug @ W1 + b1) * mk )    [1024 x 1024, K=256]
// Block tile 32x64 (8 waves, one 16x16 tile each).
//   A (32x256 bf16, 16 KB)  staged via per-lane GLOBAL_LOAD_ASYNC_TO_LDS_B128
//   B (64x256 bf16, 32 KB)  staged via one TDM tensor_load_to_lds descriptor
// ---------------------------------------------------------------------------
__global__ __launch_bounds__(256) void gemm1_kernel(
    const __bf16* __restrict__ A,   // x_aug bf16 [1024 x 256]
    const __bf16* __restrict__ Bt,  // W1t       [1024 x 256] (col-major W1)
    const float*  __restrict__ b1,
    const unsigned char* __restrict__ mk,
    __bf16* __restrict__ H)
{
    __shared__ __align__(16) __bf16 sA[32 * 256];   // 16 KB
    __shared__ __align__(16) __bf16 sB[64 * 256];   // 32 KB

    const int tid  = threadIdx.x;
    const int lane = tid & 31;
    const int bm   = blockIdx.x >> 4;               // 0..31 (M/32)
    const int bn   = blockIdx.x & 15;               // 0..15 (N/64)
    const int wm   = (tid >> 5) >> 2;               // 0..1
    const int wn   = (tid >> 5) & 3;                // 0..3

    // --- stage A tile: 1024 x 16B chunks, 4 per thread, on ASYNCcnt ---------
    #pragma unroll
    for (int it = 0; it < 4; ++it) {
        const int f = tid + it * 256;
        const int r = f >> 5;                       // row 0..31
        const int c = (f & 31) << 3;                // col (8 bf16 per 16B)
        async_copy_b128(lds_addr_of(&sA[r * 256 + c]),
                        A + (bm * 32 + r) * IDIM + c);
    }

    // --- stage B tile: one TDM 2D-tile descriptor, issued by wave 0 ---------
    if (tid < 32) {
        const unsigned long long ga = (unsigned long long)(size_t)(Bt + bn * 64 * IDIM);
        // D# group 0: count=1 | lds_addr | global_addr | type=2
        const su4 g0 = { 1u,
                         lds_addr_of(&sB[0]),
                         (unsigned)ga,
                         (unsigned)(ga >> 32) | 0x80000000u };
        // D# group 1: data_size=2B; tensor_dim0=256, tensor_dim1=64;
        //             tile_dim0=256, tile_dim1=64; dim0_stride=256
        const su8 g1 = { 0x00010000u,               // wg_mask=0 | data_size=1 (2B)
                         (256u & 0xffffu) << 16,    // tensor_dim0[15:0] @ [63:48]
                         64u << 16,                 // tensor_dim1[15:0] @ [95:80]
                         256u << 16,                // tile_dim0 @ [127:112]
                         64u,                       // tile_dim1 @ [143:128]
                         256u,                      // tensor_dim0_stride[31:0]
                         0u, 0u };
        asm volatile("tensor_load_to_lds %0, %1" :: "s"(g0), "s"(g1) : "memory");
        __builtin_amdgcn_s_wait_tensorcnt(0);
    }

    wait_async_all();
    __syncthreads();

    // --- compute: 8 K-steps of v_wmma from LDS ------------------------------
    v8f c = {0.f, 0.f, 0.f, 0.f, 0.f, 0.f, 0.f, 0.f};
    #pragma unroll
    for (int k0 = 0; k0 < IDIM; k0 += 32) {
        const v16bf a = load_fragA(sA, IDIM, wm * 16, k0, lane);
        const v16bf b = load_fragB(sB, IDIM, wn * 16, k0, lane);
        c = __builtin_amdgcn_wmma_f32_16x16x32_bf16(false, a, false, b,
                                                    (short)0, c, false, false);
    }

    const int n     = bn * 64 + wn * 16 + (lane & 15);
    const int mbase = bm * 32 + wm * 16 + ((lane >> 4) << 3);
    const float bias = b1[n];
    #pragma unroll
    for (int r = 0; r < 8; ++r) {
        const int m = mbase + r;
        const float hv = (c[r] + bias) * (float)mk[m * HDIM + n];
        H[m * HDIM + n] = (__bf16)hv;
    }
}

// ---------------------------------------------------------------------------
// Kernel 5: GEMM2 + residual + masked-MSE partial loss  [1024 x 256, K=1024]
// Block tile 32x64, K chunked by 256 through LDS (async copies + prefetch)
// ---------------------------------------------------------------------------
__global__ __launch_bounds__(256) void gemm2_loss_kernel(
    const __bf16* __restrict__ H,    // [1024 x 1024]
    const __bf16* __restrict__ Bt,   // W2t [256 x 1024]
    const float*  __restrict__ b2,
    const float*  __restrict__ xaug, // residual, f32 [1024 x 256]
    const float*  __restrict__ y,    // target,  f32 [1024 x 256]
    float* __restrict__ out,         // d_out + 1
    float* __restrict__ accum)       // [0]=sum sq err, [1]=count
{
    __shared__ __align__(16) __bf16 sA[32 * 256];   // 16 KB
    __shared__ __align__(16) __bf16 sB[64 * 256];   // 32 KB

    const int tid  = threadIdx.x;
    const int lane = tid & 31;
    const int bm   = blockIdx.x >> 2;               // 0..31 (M/32)
    const int bn   = blockIdx.x & 3;                // 0..3  (N/64)
    const int wm   = (tid >> 5) >> 2;               // 0..1
    const int wn   = (tid >> 5) & 3;                // 0..3

    v8f c = {0.f, 0.f, 0.f, 0.f, 0.f, 0.f, 0.f, 0.f};

    for (int kc = 0; kc < HDIM; kc += 256) {
        if (kc) __syncthreads();                    // LDS reuse fence

        #pragma unroll
        for (int it = 0; it < 4; ++it) {            // A chunk: 32 x 256
            const int f = tid + it * 256;
            const int r = f >> 5;
            const int cc = (f & 31) << 3;
            async_copy_b128(lds_addr_of(&sA[r * 256 + cc]),
                            H + (bm * 32 + r) * HDIM + kc + cc);
        }
        #pragma unroll
        for (int it = 0; it < 8; ++it) {            // B chunk: 64 x 256
            const int f = tid + it * 256;
            const int r = f >> 5;
            const int cc = (f & 31) << 3;
            async_copy_b128(lds_addr_of(&sB[r * 256 + cc]),
                            Bt + (bn * 64 + r) * HDIM + kc + cc);
        }
        if (kc + 256 < HDIM) {                      // pull next A chunk toward L2/L0
            __builtin_prefetch(H + (bm * 32 + (tid >> 3)) * HDIM
                                 + kc + 256 + ((tid & 7) << 5), 0, 0);
        }

        wait_async_all();
        __syncthreads();

        #pragma unroll
        for (int k0 = 0; k0 < 256; k0 += 32) {
            const v16bf a = load_fragA(sA, 256, wm * 16, k0, lane);
            const v16bf b = load_fragB(sB, 256, wn * 16, k0, lane);
            c = __builtin_amdgcn_wmma_f32_16x16x32_bf16(false, a, false, b,
                                                        (short)0, c, false, false);
        }
    }

    const int n     = bn * 64 + wn * 16 + (lane & 15);
    const int mbase = bm * 32 + wm * 16 + ((lane >> 4) << 3);
    const float bias = b2[n];
    float lsum = 0.0f, lcnt = 0.0f;
    #pragma unroll
    for (int r = 0; r < 8; ++r) {
        const int m   = mbase + r;
        const int off = m * ODIM + n;
        const float o = c[r] + bias + xaug[off];
        out[off] = o;
        const float yv = y[off];
        if (yv != 0.0f) {
            const float d = o - yv;
            lsum = fmaf(d, d, lsum);
            lcnt += 1.0f;
        }
    }
    #pragma unroll
    for (int off = 16; off > 0; off >>= 1) {        // wave32 butterfly reduce
        lsum += __shfl_xor(lsum, off, 32);
        lcnt += __shfl_xor(lcnt, off, 32);
    }
    if (lane == 0) {
        atomicAdd(&accum[0], lsum);
        atomicAdd(&accum[1], lcnt);
    }
}

__global__ void finalize_kernel(const float* __restrict__ accum,
                                float* __restrict__ loss)
{
    loss[0] = accum[0] / accum[1];
}

// ---------------------------------------------------------------------------
extern "C" void kernel_launch(void* const* d_in, const int* in_sizes, int n_in,
                              void* d_out, int out_size, void* d_ws, size_t ws_size,
                              hipStream_t stream)
{
    const float* x    = (const float*)d_in[0];   // [4,256,256]
    const float* y    = (const float*)d_in[1];   // [4,256,256]
    const float* W1   = (const float*)d_in[2];   // [256,1024]
    const float* b1   = (const float*)d_in[3];   // [1024]
    const float* W2   = (const float*)d_in[4];   // [1024,256]
    const float* b2   = (const float*)d_in[5];   // [256]
    const float* freq = (const float*)d_in[6];   // [1024]

    float*  out_full = (float*)d_out;            // [0]=loss, [1..]=out (1024x256)

    char* ws = (char*)d_ws;
    float*  xaug_f  = (float*) (ws + 0);                                  // 1 MB
    __bf16* xaug_bf = (__bf16*)(ws + (1u << 20));                         // 512 KB
    __bf16* W1t     = (__bf16*)(ws + (1u << 20) + (1u << 19));            // 512 KB
    __bf16* W2t     = (__bf16*)(ws + (1u << 20) + (2u << 19));            // 512 KB
    __bf16* H       = (__bf16*)(ws + (1u << 20) + (3u << 19));            // 2 MB
    unsigned char* mk = (unsigned char*)(ws + (1u << 20) + (3u << 19) + (1u << 21)); // 1 MB
    float* accum    = (float*) (ws + (1u << 20) + (3u << 19) + (1u << 21) + (1u << 20));

    hipMemsetAsync(accum, 0, 2 * sizeof(float), stream);

    convert_weights_kernel<<<(2 * IDIM * HDIM) / 256, 256, 0, stream>>>(W1, W2, W1t, W2t);
    mask_kernel<<<(BT * HDIM / 2) / 256, 256, 0, stream>>>(freq, mk);
    argaug_kernel<<<BT, 256, 0, stream>>>(x, y, xaug_f, xaug_bf);

    // GEMM1: grid = (1024/32) x (1024/64) = 32 x 16 = 512 blocks
    gemm1_kernel<<<512, 256, 0, stream>>>(xaug_bf, W1t, b1, mk, H);
    // GEMM2: grid = (1024/32) x (256/64) = 32 x 4 = 128 blocks
    gemm2_loss_kernel<<<128, 256, 0, stream>>>(H, W2t, b2, xaug_f, y,
                                               out_full + 1, accum);
    finalize_kernel<<<1, 1, 0, stream>>>(accum, out_full);
}
